// MemoryRamTwoStreamModule_batch_25623774888601
// MI455X (gfx1250) — matrix operand order repaired and checked by reading, not compiled, not run on hardware
//
#include <hip/hip_runtime.h>

// ---------------------------------------------------------------------------
// MI455X (gfx1250) implementation of the two-stream memory RNN scan.
// All heavy GEMMs use v_wmma_f32_16x16x32_bf16 (bf16 inputs, f32 accumulate).
// Weights are pre-converted to bf16 once per call, so weight staging is pure
// aligned copies; activations are converted with v_cvt_pk_bf16_f32 when the
// builtin is available. Block tile 32x128, K-chunk 64 -> 4 WMMAs/barrier pair.
// ---------------------------------------------------------------------------

typedef __attribute__((ext_vector_type(16))) __bf16       v16bf;
typedef __attribute__((ext_vector_type(8)))  float        v8f;
typedef __attribute__((ext_vector_type(8)))  unsigned int v8u;

constexpr int Bsz = 128;   // batch
constexpr int Tn  = 128;   // timesteps
constexpr int Dn  = 512;   // input size
constexpr int Hn  = 512;   // hidden
constexpr int Mn  = 100;   // memory slots
constexpr int Mp  = 112;   // Mn padded to multiple of 16

#if defined(__has_builtin)
#if __has_builtin(__builtin_amdgcn_cvt_pk_bf16_f32)
#define HAVE_CVT_PK_BF16 1
#endif
#endif

// round-to-nearest-even f32 -> bf16, packed as two bf16 in one dword
__device__ __forceinline__ unsigned int pack2bf(float x, float y) {
#ifdef HAVE_CVT_PK_BF16
    auto p = __builtin_amdgcn_cvt_pk_bf16_f32(x, y);
    return __builtin_bit_cast(unsigned int, p);
#else
    unsigned int xu = __float_as_uint(x), yu = __float_as_uint(y);
    xu += 0x7FFFu + ((xu >> 16) & 1u);
    yu += 0x7FFFu + ((yu >> 16) & 1u);
    return (xu >> 16) | (yu & 0xFFFF0000u);
#endif
}

// ---------------------------------------------------------------------------
// Generic GEMM: C[B x N] = act( concat(A0,A1,A2)[B x Ktot] @ W^T + bias )
// W16 = bf16-packed weights, [N][Ktot/2] dwords row-major.
// Segments K0/K1/K2 are multiples of 64. N must be a multiple of 128.
// Block: 256 threads = 8 wave32s; 32x128 C tile; each wave computes two
// 16x16 tiles (cols c and c+64) sharing one A fragment.
// ---------------------------------------------------------------------------
__global__ __launch_bounds__(256) void gemm3_relu_kernel(
    const float* __restrict__ A0, int ldA0, int K0,
    const float* __restrict__ A1, int ldA1, int K1,
    const float* __restrict__ A2, int ldA2, int K2,
    const unsigned int* __restrict__ W16, const float* __restrict__ bias,
    int Ktot,
    float* __restrict__ C, int ldC,
    float* __restrict__ C2, int ldC2, int do_relu) {
    __shared__ unsigned int sA[32 * 34];   // 32 rows x 32 k-pairs (+2 pad)
    __shared__ unsigned int sB[128 * 34];  // 128 cols x 32 k-pairs (+2 pad)

    const int tid  = threadIdx.x;
    const int wave = tid >> 5;
    const int lane = tid & 31;
    const int rowTile = (wave >> 2) * 16;      // 0 or 16
    const int colT0   = (wave & 3) * 16;       // 0,16,32,48 (+64 for 2nd tile)
    const int gRow0   = blockIdx.y * 32;
    const int gCol0   = blockIdx.x * 128;

    // loop-invariant staging descriptors
    const int kp  = tid & 31;          // A: k-pair index within 64-wide chunk
    const int rA0 = tid >> 5;          // A rows rA0 + 8i (i<4)
    const int wp  = (tid & 15) * 2;    // B: dword-pair base within 32-dword chunk
    const int nB0 = tid >> 4;          // B cols nB0 + 16i (i<8)
    const int Kw  = Ktot >> 1;
    const unsigned int* __restrict__ pW0 = W16 + (size_t)(gCol0 + nB0) * Kw + wp;
    const size_t wStep = (size_t)16 * Kw;

    v8f acc0 = {}, acc1 = {};

    const float* segA[3] = { A0, A1, A2 };
    const int    segL[3] = { ldA0, ldA1, ldA2 };
    const int    segK[3] = { K0, K1, K2 };
    int kBase = 0;
    for (int s = 0; s < 3; ++s) {
        const int Ks = segK[s];
        if (Ks > 0) {
            const size_t aStep = (size_t)8 * segL[s];
            const float* __restrict__ pA =
                segA[s] + (size_t)(gRow0 + rA0) * segL[s] + kp * 2;
            const unsigned int* __restrict__ pWs = pW0 + (kBase >> 1);
            for (int kc = 0; kc < Ks; kc += 64) {
                // stage A chunk (32x64 f32 -> bf16)
#pragma unroll
                for (int i = 0; i < 4; ++i) {
                    const float* src = pA + (size_t)i * aStep;
                    sA[(rA0 + 8 * i) * 34 + kp] = pack2bf(src[0], src[1]);
                }
                // stage W chunk (128x64, already bf16): pure aligned copies
#pragma unroll
                for (int i = 0; i < 8; ++i) {
                    const uint2 v = *(const uint2*)(pWs + (size_t)i * wStep);
                    *(uint2*)&sB[(nB0 + 16 * i) * 34 + wp] = v;
                }
                __syncthreads();

                const int r   = lane & 15;
                const int aHi = (lane & 16) ? 4 : 0;  // lanes 16-31: K pairs {8..15,24..31}
                const int bHi = (lane & 16) ? 4 : 0;  // lanes 16-31: K pairs {8..15,24..31}
#pragma unroll
                for (int sh = 0; sh < 2; ++sh) {      // two 32-deep sub-chunks
                    v8u au, bu0, bu1;
#pragma unroll
                    for (int j = 0; j < 8; ++j) {
                        const int wa = (j < 4 ? j : j + 4) + aHi + sh * 16;
                        const int wb = (j < 4 ? j : j + 4) + bHi + sh * 16;
                        au[j]  = sA[(rowTile + r) * 34 + wa];
                        bu0[j] = sB[(colT0 + r) * 34 + wb];
                        bu1[j] = sB[(colT0 + 64 + r) * 34 + wb];
                    }
                    v16bf a  = __builtin_bit_cast(v16bf, au);
                    v16bf b0 = __builtin_bit_cast(v16bf, bu0);
                    v16bf b1 = __builtin_bit_cast(v16bf, bu1);
                    acc0 = __builtin_amdgcn_wmma_f32_16x16x32_bf16(
                        false, a, false, b0, (short)0, acc0, false, false);
                    acc1 = __builtin_amdgcn_wmma_f32_16x16x32_bf16(
                        false, a, false, b1, (short)0, acc1, false, false);
                }
                __syncthreads();
                pA  += 64;
                pWs += 32;
            }
        }
        kBase += Ks;
    }

    // epilogue: C VGPR layout -> row = i + 8*(lane>=16), col = lane&15
    const int r     = lane & 15;
    const int rhi   = (lane & 16) ? 8 : 0;
    const int gcol0 = gCol0 + colT0 + r;
    const int gcol1 = gcol0 + 64;
    const float bv0 = bias[gcol0], bv1 = bias[gcol1];
#pragma unroll
    for (int i = 0; i < 8; ++i) {
        const int grow = gRow0 + rowTile + rhi + i;
        float v0 = acc0[i] + bv0;
        float v1 = acc1[i] + bv1;
        if (do_relu) { v0 = fmaxf(v0, 0.f); v1 = fmaxf(v1, 0.f); }
        C[(size_t)grow * ldC + gcol0] = v0;
        C[(size_t)grow * ldC + gcol1] = v1;
        if (C2) {
            C2[(size_t)grow * ldC2 + gcol0] = v0;
            C2[(size_t)grow * ldC2 + gcol1] = v1;
        }
    }
}

// ---------------------------------------------------------------------------
// GEMM + row softmax for the M=100 heads (N padded to 112 = 7 col tiles).
// Block: 256 threads = 8 waves; waves 0-6 own one col tile each (wave 7 only
// stages; its branch around the WMMA is wave-uniform so EXEC stays all-ones
// inside every WMMA-executing wave). One block owns a full 16x112 strip, so
// the softmax is block-local. Padded cols are written as exact zeros.
// ---------------------------------------------------------------------------
__global__ __launch_bounds__(256) void gemm3_softmax_kernel(
    const float* __restrict__ A0, int ldA0, int K0,
    const float* __restrict__ A1, int ldA1, int K1,
    const float* __restrict__ A2, int ldA2, int K2,
    const unsigned int* __restrict__ W16, const float* __restrict__ bias,
    int Nreal, int Ktot,
    float* __restrict__ Out /* [B x Mp] */) {
    __shared__ unsigned int sA[16 * 34];
    __shared__ unsigned int sB[112 * 34];
    __shared__ float        sL[16 * 120];

    const int tid  = threadIdx.x;
    const int wave = tid >> 5;
    const int lane = tid & 31;
    const int gRow0 = blockIdx.x * 16;

    const int kp  = tid & 31;
    const int rA0 = tid >> 5;          // A rows rA0 + 8i (i<2)
    const int wp  = (tid & 15) * 2;    // B dword-pair base
    const int nB0 = tid >> 4;          // B cols nB0 + 16i (i<7), max 111
    const int Kw  = Ktot >> 1;
    const unsigned int* __restrict__ pW0 = W16 + (size_t)nB0 * Kw + wp;
    const size_t wStep = (size_t)16 * Kw;

    v8f acc = {};

    const float* segA[3] = { A0, A1, A2 };
    const int    segL[3] = { ldA0, ldA1, ldA2 };
    const int    segK[3] = { K0, K1, K2 };
    int kBase = 0;
    for (int s = 0; s < 3; ++s) {
        const int Ks = segK[s];
        if (Ks > 0) {
            const size_t aStep = (size_t)8 * segL[s];
            const float* __restrict__ pA =
                segA[s] + (size_t)(gRow0 + rA0) * segL[s] + kp * 2;
            const unsigned int* __restrict__ pWs = pW0 + (kBase >> 1);
            for (int kc = 0; kc < Ks; kc += 64) {
#pragma unroll
                for (int i = 0; i < 2; ++i) {               // 16x64 A chunk
                    const float* src = pA + (size_t)i * aStep;
                    sA[(rA0 + 8 * i) * 34 + kp] = pack2bf(src[0], src[1]);
                }
#pragma unroll
                for (int i = 0; i < 7; ++i) {               // 112x64 W chunk
                    const int n = nB0 + 16 * i;
                    uint2 v = make_uint2(0u, 0u);
                    if (n < Nreal) v = *(const uint2*)(pWs + (size_t)i * wStep);
                    *(uint2*)&sB[n * 34 + wp] = v;
                }
                __syncthreads();

                if (wave < 7) {
                    const int r   = lane & 15;
                    const int aHi = (lane & 16) ? 4 : 0;
                    const int bHi = (lane & 16) ? 4 : 0;
#pragma unroll
                    for (int sh = 0; sh < 2; ++sh) {
                        v8u au, bu;
#pragma unroll
                        for (int j = 0; j < 8; ++j) {
                            const int wa = (j < 4 ? j : j + 4) + aHi + sh * 16;
                            const int wb = (j < 4 ? j : j + 4) + bHi + sh * 16;
                            au[j] = sA[r * 34 + wa];
                            bu[j] = sB[(wave * 16 + r) * 34 + wb];
                        }
                        v16bf a = __builtin_bit_cast(v16bf, au);
                        v16bf b = __builtin_bit_cast(v16bf, bu);
                        acc = __builtin_amdgcn_wmma_f32_16x16x32_bf16(
                            false, a, false, b, (short)0, acc, false, false);
                    }
                }
                __syncthreads();
                pA  += 64;
                pWs += 32;
            }
        }
        kBase += Ks;
    }

    // dump logits (+bias, mask pads) to LDS
    if (wave < 7) {
        const int r   = lane & 15;
        const int col = wave * 16 + r;
        const int rhi = (lane & 16) ? 8 : 0;
        const float bv = (col < Nreal) ? bias[col] : 0.f;
#pragma unroll
        for (int i = 0; i < 8; ++i) {
            float v = acc[i] + bv;
            if (col >= Nreal) v = -1e30f;
            sL[(rhi + i) * 120 + col] = v;
        }
    }
    __syncthreads();

    // block-local masked softmax, 16 rows handled by threads 0..15
    if (tid < 16) {
        const int row = tid;
        float mx = -1e30f;
        for (int c = 0; c < Mp; ++c) mx = fmaxf(mx, sL[row * 120 + c]);
        float ssum = 0.f;
        for (int c = 0; c < Nreal; ++c) {
            float e = __expf(sL[row * 120 + c] - mx);
            sL[row * 120 + c] = e;
            ssum += e;
        }
        const float inv = 1.f / ssum;
        for (int c = 0; c < Mp; ++c)
            Out[(size_t)(gRow0 + row) * Mp + c] =
                (c < Nreal) ? sL[row * 120 + c] * inv : 0.f;
    }
}

// ---------------------------------------------------------------------------
// 3-way write gate: aw = softmax([ha,hm,h] @ W_wp^T + b_wp), W_wp: [3 x 1536]
// ---------------------------------------------------------------------------
__global__ __launch_bounds__(128) void aw_softmax_kernel(
    const float* __restrict__ ha, const float* __restrict__ hm,
    const float* __restrict__ h,
    const float* __restrict__ W, const float* __restrict__ bias,
    float* __restrict__ aw /* [B x 4] */) {
    const int b = threadIdx.x;
    float a0 = bias[0], a1 = bias[1], a2 = bias[2];
    const float* segs[3] = { ha, hm, h };
    for (int s = 0; s < 3; ++s) {
        const float* x = segs[s] + (size_t)b * Hn;
        const float* w = W + s * Hn;
        for (int k = 0; k < Hn; ++k) {
            const float xv = x[k];
            a0 += xv * w[k];
            a1 += xv * w[k + 1536];
            a2 += xv * w[k + 2 * 1536];
        }
    }
    const float mx = fmaxf(a0, fmaxf(a1, a2));
    const float e0 = __expf(a0 - mx), e1 = __expf(a1 - mx), e2 = __expf(a2 - mx);
    const float inv = 1.f / (e0 + e1 + e2);
    aw[b * 4 + 0] = e0 * inv;
    aw[b * 4 + 1] = e1 * inv;
    aw[b * 4 + 2] = e2 * inv;
}

// ---------------------------------------------------------------------------
// Fused memory pass: one block per batch element b.
//   r[b,h]     = sum_m ar[b,m] * mem_old[b,m,h]
//   mem[b,m,h] = aw0*mem_old + aw1*awa[b,m]*cta[b,h] + aw2*awm[b,m]*ctm[b,h]
// Single read-modify-write sweep over mem[b] (L2-resident, 400KB/b per step).
// ---------------------------------------------------------------------------
__global__ __launch_bounds__(256) void mem_r_kernel(
    float* __restrict__ mem,
    const float* __restrict__ ar, const float* __restrict__ aw,
    const float* __restrict__ awa, const float* __restrict__ awm,
    const float* __restrict__ cta, const float* __restrict__ ctm,
    float* __restrict__ r) {
    const int b = blockIdx.x;
    const int t = threadIdx.x;
    __shared__ float s_ar[Mn], s_fa[Mn], s_fm[Mn];

    const float aw0 = aw[b * 4 + 0];
    const float aw1 = aw[b * 4 + 1];
    const float aw2 = aw[b * 4 + 2];
    for (int m = t; m < Mn; m += 256) {
        s_ar[m] = ar[(size_t)b * Mp + m];
        s_fa[m] = aw1 * awa[(size_t)b * Mp + m];
        s_fm[m] = aw2 * awm[(size_t)b * Mp + m];
    }
    __syncthreads();

    const int h0 = t, h1 = t + 256;
    const float ca0 = cta[(size_t)b * Hn + h0], ca1 = cta[(size_t)b * Hn + h1];
    const float cm0 = ctm[(size_t)b * Hn + h0], cm1 = ctm[(size_t)b * Hn + h1];
    float r0 = 0.f, r1 = 0.f;
    float* mb = mem + (size_t)b * Mn * Hn;
    for (int m = 0; m < Mn; ++m) {
        const float v0 = mb[m * Hn + h0];
        const float v1 = mb[m * Hn + h1];
        r0 += s_ar[m] * v0;
        r1 += s_ar[m] * v1;
        mb[m * Hn + h0] = aw0 * v0 + s_fa[m] * ca0 + s_fm[m] * cm0;
        mb[m * Hn + h1] = aw0 * v1 + s_fa[m] * ca1 + s_fm[m] * cm1;
    }
    r[(size_t)b * Hn + h0] = r0;
    r[(size_t)b * Hn + h1] = r1;
}

// f32 -> packed bf16 pairs (one dword per pair), grid-stride
__global__ void f32_to_bf16_kernel(const float* __restrict__ src,
                                   unsigned int* __restrict__ dst, long npairs) {
    long i = (long)blockIdx.x * blockDim.x + threadIdx.x;
    const long stride = (long)gridDim.x * blockDim.x;
    for (; i < npairs; i += stride)
        dst[i] = pack2bf(src[2 * i], src[2 * i + 1]);
}

__global__ void zero_kernel(float* p, long n) {
    long i = (long)blockIdx.x * blockDim.x + threadIdx.x;
    const long stride = (long)gridDim.x * blockDim.x;
    for (; i < n; i += stride) p[i] = 0.f;
}

// ---------------------------------------------------------------------------
// Host side
// ---------------------------------------------------------------------------
extern "C" void kernel_launch(void* const* d_in, const int* in_sizes, int n_in,
                              void* d_out, int out_size, void* d_ws, size_t ws_size,
                              hipStream_t stream) {
    (void)in_sizes; (void)n_in; (void)out_size; (void)ws_size;

    const float* XA    = (const float*)d_in[0];   // [B,T,D]
    const float* XM    = (const float*)d_in[1];   // [B,T,D]
    // d_in[2] = nImg (int scalar) — shapes are compile-time constants here.
    const float* W_ca  = (const float*)d_in[3];
    const float* b_ca  = (const float*)d_in[4];
    const float* W_cm  = (const float*)d_in[5];
    const float* b_cm  = (const float*)d_in[6];
    const float* W_wp  = (const float*)d_in[7];
    const float* b_wp  = (const float*)d_in[8];
    const float* W_wpa = (const float*)d_in[9];
    const float* b_wpa = (const float*)d_in[10];
    const float* W_wpm = (const float*)d_in[11];
    const float* b_wpm = (const float*)d_in[12];
    const float* W_rp  = (const float*)d_in[13];
    const float* b_rp  = (const float*)d_in[14];
    const float* W_rh  = (const float*)d_in[15];
    const float* b_rh  = (const float*)d_in[16];
    const float* W_rha = (const float*)d_in[17];
    const float* b_rha = (const float*)d_in[18];
    const float* W_rhm = (const float*)d_in[19];
    const float* b_rhm = (const float*)d_in[20];

    float* out = (float*)d_out;   // [B,T,H]
    float* ws  = (float*)d_ws;

    size_t off = 0;
    float* mem = ws + off;              off += (size_t)Bsz * Mn * Hn;  // 6.55M
    float* st[2][3];
    for (int s = 0; s < 2; ++s)
        for (int j = 0; j < 3; ++j) { st[s][j] = ws + off; off += (size_t)Bsz * Hn; }
    float* rbuf = ws + off;             off += (size_t)Bsz * Hn;
    float* ar   = ws + off;             off += (size_t)Bsz * Mp;
    float* awa  = ws + off;             off += (size_t)Bsz * Mp;
    float* awm  = ws + off;             off += (size_t)Bsz * Mp;
    float* aw   = ws + off;             off += (size_t)Bsz * 4;
    float* cta  = ws + off;             off += (size_t)Bsz * Hn;
    float* ctm  = ws + off;             off += (size_t)Bsz * Hn;

    // bf16 weight caches (dword = 2 packed bf16), 8-byte aligned
    unsigned int* wb = (unsigned int*)(ws + off);
    size_t uoff = 0;
    unsigned int* wb_ca  = wb + uoff;   uoff += (size_t)Hn * (Hn + Dn) / 2;
    unsigned int* wb_cm  = wb + uoff;   uoff += (size_t)Hn * (Hn + Dn) / 2;
    unsigned int* wb_rh  = wb + uoff;   uoff += (size_t)Hn * (2 * Hn) / 2;
    unsigned int* wb_rha = wb + uoff;   uoff += (size_t)Hn * (Dn + 2 * Hn) / 2;
    unsigned int* wb_rhm = wb + uoff;   uoff += (size_t)Hn * (Dn + 2 * Hn) / 2;
    unsigned int* wb_rp  = wb + uoff;   uoff += (size_t)Mn * (3 * Hn) / 2;
    unsigned int* wb_wpa = wb + uoff;   uoff += (size_t)Mn * (Hn + Dn) / 2;
    unsigned int* wb_wpm = wb + uoff;   uoff += (size_t)Mn * (Hn + Dn) / 2;

    // one-time (per call) weight conversion f32 -> bf16
    f32_to_bf16_kernel<<<512, 256, 0, stream>>>(W_ca,  wb_ca,  (long)Hn * (Hn + Dn) / 2);
    f32_to_bf16_kernel<<<512, 256, 0, stream>>>(W_cm,  wb_cm,  (long)Hn * (Hn + Dn) / 2);
    f32_to_bf16_kernel<<<512, 256, 0, stream>>>(W_rh,  wb_rh,  (long)Hn * (2 * Hn) / 2);
    f32_to_bf16_kernel<<<512, 256, 0, stream>>>(W_rha, wb_rha, (long)Hn * (Dn + 2 * Hn) / 2);
    f32_to_bf16_kernel<<<512, 256, 0, stream>>>(W_rhm, wb_rhm, (long)Hn * (Dn + 2 * Hn) / 2);
    f32_to_bf16_kernel<<<512, 256, 0, stream>>>(W_rp,  wb_rp,  (long)Mn * (3 * Hn) / 2);
    f32_to_bf16_kernel<<<512, 256, 0, stream>>>(W_wpa, wb_wpa, (long)Mn * (Hn + Dn) / 2);
    f32_to_bf16_kernel<<<512, 256, 0, stream>>>(W_wpm, wb_wpm, (long)Mn * (Hn + Dn) / 2);

    // zero mem + state set 0 (contiguous at the front of ws)
    const long zero_n = (long)Bsz * Mn * Hn + 3L * Bsz * Hn;
    zero_kernel<<<2048, 256, 0, stream>>>(ws, zero_n);

    const dim3 gH(Hn / 128, Bsz / 32);  // (4,4) for N=512 GEMMs
    const int  ldX = Tn * Dn;           // row stride of [B,T,D] time slices

    for (int t = 0; t < Tn; ++t) {
        const int c = t & 1, n = c ^ 1;
        const float* h  = st[c][0];
        const float* ha = st[c][1];
        const float* hm = st[c][2];
        float* h_n  = st[n][0];
        float* ha_n = st[n][1];
        float* hm_n = st[n][2];
        const float* xa_t = XA + (size_t)t * Dn;
        const float* xm_t = XM + (size_t)t * Dn;

        // read attention: ar = softmax([ha,hm,h] @ W_rp^T + b_rp)
        gemm3_softmax_kernel<<<Bsz / 16, 256, 0, stream>>>(
            ha, Hn, Hn, hm, Hn, Hn, h, Hn, Hn,
            wb_rp, b_rp, Mn, 3 * Hn, ar);
        // write attentions over the streams
        gemm3_softmax_kernel<<<Bsz / 16, 256, 0, stream>>>(
            ha, Hn, Hn, xa_t, ldX, Dn, nullptr, 0, 0,
            wb_wpa, b_wpa, Mn, Hn + Dn, awa);
        gemm3_softmax_kernel<<<Bsz / 16, 256, 0, stream>>>(
            hm, Hn, Hn, xm_t, ldX, Dn, nullptr, 0, 0,
            wb_wpm, b_wpm, Mn, Hn + Dn, awm);
        // 3-way gate
        aw_softmax_kernel<<<1, 128, 0, stream>>>(ha, hm, h, W_wp, b_wp, aw);
        // content vectors
        gemm3_relu_kernel<<<gH, 256, 0, stream>>>(
            ha, Hn, Hn, xa_t, ldX, Dn, nullptr, 0, 0,
            wb_ca, b_ca, Hn + Dn, cta, Hn, nullptr, 0, 1);
        gemm3_relu_kernel<<<gH, 256, 0, stream>>>(
            hm, Hn, Hn, xm_t, ldX, Dn, nullptr, 0, 0,
            wb_cm, b_cm, Hn + Dn, ctm, Hn, nullptr, 0, 1);
        // fused read + memory update (uses old mem for r, writes new mem)
        mem_r_kernel<<<Bsz, 256, 0, stream>>>(mem, ar, aw, awa, awm, cta, ctm, rbuf);
        // next hidden states; h1 also streams into out[:, t, :]
        gemm3_relu_kernel<<<gH, 256, 0, stream>>>(
            rbuf, Hn, Hn, h, Hn, Hn, nullptr, 0, 0,
            wb_rh, b_rh, 2 * Hn, h_n, Hn,
            out + (size_t)t * Hn, Tn * Hn, 1);
        gemm3_relu_kernel<<<gH, 256, 0, stream>>>(
            xa_t, ldX, Dn, rbuf, Hn, Hn, ha, Hn, Hn,
            wb_rha, b_rha, Dn + 2 * Hn, ha_n, Hn, nullptr, 0, 1);
        gemm3_relu_kernel<<<gH, 256, 0, stream>>>(
            xm_t, ldX, Dn, rbuf, Hn, Hn, hm, Hn, Hn,
            wb_rhm, b_rhm, Dn + 2 * Hn, hm_n, Hn, nullptr, 0, 1);
    }
}
